// OffloadedModel_41102837023419
// MI455X (gfx1250) — matrix operand order repaired and checked by reading, not compiled, not run on hardware
//
#include <hip/hip_runtime.h>
#include <hip/hip_bf16.h>
#include <stdint.h>

// ---------------------------------------------------------------------------
// MoE (router top-2 + MXFP4 SwiGLU experts) for MI455X / gfx1250.
// Bandwidth-bound: ~215MB of int32-packed fp4 weight bytes dominate
// (~9us @ 23.3TB/s). Weights stream once with NT b128 loads; activations
// are staged into double-buffered LDS with GLOBAL_LOAD_ASYNC_TO_LDS_B128
// (ASYNCcnt path, copy overlapped with WMMA); fp4->f16 dequant uses
// v_perm_b32 byte LUTs (all e2m1 f16 magnitudes have low byte 0x00) +
// exact v_pk_mul_f16 e8m0 scaling; GEMMs use v_wmma_f32_16x16x32_f16
// with M=128 token tiles so every weight byte is read exactly once.
// ---------------------------------------------------------------------------

typedef __attribute__((ext_vector_type(16))) _Float16 v16h;
typedef __attribute__((ext_vector_type(8)))  _Float16 v8h;
typedef __attribute__((ext_vector_type(2)))  _Float16 v2h;
typedef __attribute__((ext_vector_type(8)))  float    v8f;
typedef __attribute__((ext_vector_type(4)))  int      v4i;

#define NE   16      // experts
#define HD   1024    // hidden
#define FD   2048    // ffn
#define NT   128     // tokens (B*S)
#define APAD 48      // LDS row stride in halves (pad to kill bank conflicts)

union FragU {
  v16h     v;
  uint32_t u[8];
  v2h      h2[8];
};

// ---- dequant 8 packed-fp4 bytes held in one u32 ---------------------------
// w byte j = packed byte for elements 2j (low nibble), 2j+1 (high nibble).
__device__ __forceinline__ void deq8(uint32_t w, uint32_t* out) {
  const uint32_t lutLO = 0x3E3C3800u;   // f16 high bytes: 0, .5, 1, 1.5
  const uint32_t lutHI = 0x46444240u;   //                 2, 3, 4, 6
  uint32_t mlo = w & 0x07070707u;               // low-nibble magnitudes
  uint32_t mhi = (w >> 4) & 0x07070707u;        // high-nibble magnitudes
  uint32_t slo = (w << 4) & 0x80808080u;        // low-nibble signs -> bit7
  uint32_t shi = w & 0x80808080u;               // high-nibble signs -> bit7
  uint32_t hbl = __builtin_amdgcn_perm(lutHI, lutLO, mlo) | slo;
  uint32_t hbh = __builtin_amdgcn_perm(lutHI, lutLO, mhi) | shi;
  // expand: pair k -> bytes {0x00, hbl_k, 0x00, hbh_k}  (sel 0x0C == zero)
  out[0] = __builtin_amdgcn_perm(hbh, hbl, 0x040C000Cu);
  out[1] = __builtin_amdgcn_perm(hbh, hbl, 0x050C010Cu);
  out[2] = __builtin_amdgcn_perm(hbh, hbl, 0x060C020Cu);
  out[3] = __builtin_amdgcn_perm(hbh, hbl, 0x070C030Cu);
}

// ---- B-matrix fragment (32x16, 16-bit): lane n = lane&15 holds
// K = kb + [0..7] and kb + 16 + [0..7], kb = k0 + 8*(lane>>4). One e8m0 scale per 32-K block.
__device__ __forceinline__ v16h load_wfrag(const int* __restrict__ brow,
                                           const int* __restrict__ srow,
                                           int k0, int lane) {
  const int kb = k0 + ((lane >> 4) << 3);
  const v4i* p = (const v4i*)(brow + (kb >> 1));
  v4i q0 = __builtin_nontemporal_load(p);       // 8 packed bytes, streamed (NT)
  v4i q1 = __builtin_nontemporal_load(p + 2);
  int s = srow[k0 >> 5];
  // 2^(s-127) as f16 bits: exponent field = s-127+15 (exact for s in [113,142])
  v2h sc2;
  sc2[0] = __builtin_bit_cast(_Float16, (unsigned short)((s - 112) << 10));
  sc2[1] = sc2[0];
  uint32_t w0 = (uint32_t)q0.x | ((uint32_t)q0.y << 8) |
                ((uint32_t)q0.z << 16) | ((uint32_t)q0.w << 24);
  uint32_t w1 = (uint32_t)q1.x | ((uint32_t)q1.y << 8) |
                ((uint32_t)q1.z << 16) | ((uint32_t)q1.w << 24);
  FragU f;
  deq8(w0, &f.u[0]);
  deq8(w1, &f.u[4]);
#pragma unroll
  for (int j = 0; j < 8; ++j) f.h2[j] = f.h2[j] * sc2;  // v_pk_mul_f16, exact
  return f.v;
}

// ---- A-matrix fragment (16x32, 16-bit) from LDS: lane m = lane&15,
// K = khalf + [0..7] and khalf + 16 + [0..7], khalf = 8*(lane>>4). ----
__device__ __forceinline__ v16h load_afrag(const _Float16* p) {
  v8h a0 = *(const v8h*)p;          // ds_load_b128
  v8h a1 = *(const v8h*)(p + 16);   // ds_load_b128
  v16h a;
#pragma unroll
  for (int i = 0; i < 8; ++i) { a[i] = a0[i]; a[i + 8] = a1[i]; }
  return a;
}

// ---- async copy of one 32B-per-thread A slice into LDS --------------------
// INST_OFFSET is added to BOTH the LDS and global addresses (ISA 10.x), so
// offset:16 covers the second 16B half of each thread's 32B slice.
__device__ __forceinline__ void issue_async(uint32_t ldsoff, const void* g) {
  asm volatile(
      "global_load_async_to_lds_b128 %0, %1, off\n\t"
      "global_load_async_to_lds_b128 %0, %1, off offset:16"
      :: "v"(ldsoff), "v"(g) : "memory");
}
__device__ __forceinline__ void wait_async0() {
  asm volatile("s_wait_asynccnt 0x0" ::: "memory");
}

// ---------------------------------------------------------------------------
// Kernel 1: fp32 -> f16 activation convert
// ---------------------------------------------------------------------------
__global__ void cvt_f32_f16(const float* __restrict__ X,
                            _Float16* __restrict__ Xh, int n) {
  int i = blockIdx.x * blockDim.x + threadIdx.x;
  if (i < n) Xh[i] = (_Float16)X[i];
}

// ---------------------------------------------------------------------------
// Kernel 2: router logits + top-2 + softmax -> dense combine[T, NE]
// ---------------------------------------------------------------------------
__global__ __launch_bounds__(256) void router_topk(
    const float* __restrict__ X, const float* __restrict__ RW,
    float* __restrict__ combine) {
  __shared__ float logits[NE];
  const int t = blockIdx.x;
  const int wave = threadIdx.x >> 5;
  const int lane = threadIdx.x & 31;
#pragma unroll
  for (int i = 0; i < 2; ++i) {
    const int e = wave * 2 + i;
    float s = 0.f;
    for (int h = lane; h < HD; h += 32)
      s += X[(size_t)t * HD + h] * RW[(size_t)e * HD + h];
#pragma unroll
    for (int off = 16; off > 0; off >>= 1)
      s += __shfl_xor(s, off, 32);
    if (lane == 0) logits[e] = s;
  }
  __syncthreads();
  if (threadIdx.x == 0) {
    int i1 = 0; float v1 = logits[0];
    for (int e = 1; e < NE; ++e)
      if (logits[e] > v1) { v1 = logits[e]; i1 = e; }
    int i2 = -1; float v2 = -3.4e38f;
    for (int e = 0; e < NE; ++e)
      if (e != i1 && logits[e] > v2) { v2 = logits[e]; i2 = e; }
    const float w2e = __expf(v2 - v1);          // softmax over {v1, v2}
    const float inv = 1.0f / (1.0f + w2e);
    for (int e = 0; e < NE; ++e) {
      float w = (e == i1) ? inv : (e == i2) ? w2e * inv : 0.0f;
      combine[t * NE + e] = w;
    }
  }
}

// ---------------------------------------------------------------------------
// Kernel 3: gu = X @ Wgu^T + bias; h = silu(gate)*up  (f16 out to ws)
// grid (E, F/128); 8 waves; wave owns 16 gate + 16 up cols x 128 tokens.
// Double-buffered async LDS staging: copy of chunk i+1 overlaps WMMA of i.
// ---------------------------------------------------------------------------
__global__ __launch_bounds__(256) void moe_gu_swiglu(
    const _Float16* __restrict__ Xh,
    const int* __restrict__ blocks_gu,
    const int* __restrict__ scales_gu,
    const float* __restrict__ bias_gu,
    _Float16* __restrict__ Hbuf) {
  __shared__ _Float16 Atile[2][NT][APAD];
  const int e     = blockIdx.x;
  const int ftile = blockIdx.y * 128;
  const int wave  = threadIdx.x >> 5;
  const int lane  = threadIdx.x & 31;
  const int ncol  = lane & 15;
  const int khalf = (lane >> 4) << 3;
  const int og = ftile + wave * 16 + ncol;   // gate column in [0, F)
  const int ou = og + FD;                    // matching up column
  const int* browg = blocks_gu + ((size_t)e * 2 * FD + og) * (HD / 2);
  const int* browu = blocks_gu + ((size_t)e * 2 * FD + ou) * (HD / 2);
  const int* srowg = scales_gu + ((size_t)e * 2 * FD + og) * (HD / 32);
  const int* srowu = scales_gu + ((size_t)e * 2 * FD + ou) * (HD / 32);

  v8f accg[8] = {};
  v8f accu[8] = {};

  const int srow_r = threadIdx.x >> 1;        // staging row
  const int scol   = (threadIdx.x & 1) * 16;  // staging half-offset
  const uint32_t ldsoff0 = (uint32_t)(uintptr_t)&Atile[0][srow_r][scol];
  const uint32_t ldsoff1 = (uint32_t)(uintptr_t)&Atile[1][srow_r][scol];
  const _Float16* Asrc = Xh + (size_t)srow_r * HD + scol;

  issue_async(ldsoff0, (const void*)Asrc);
  wait_async0();
  __syncthreads();

  const int NK = HD / 32;
  for (int i = 0; i < NK; ++i) {
    const int k0 = i * 32;
    if (i + 1 < NK)   // prefetch next A chunk into the other buffer (async)
      issue_async((i & 1) ? ldsoff0 : ldsoff1, (const void*)(Asrc + (i + 1) * 32));
    if (k0 + 32 < HD) {
      __builtin_prefetch(browg + ((k0 + 32) >> 1), 0, 1);
      __builtin_prefetch(browu + ((k0 + 32) >> 1), 0, 1);
    }
    const _Float16 (*At)[APAD] = Atile[i & 1];
    v16h bg = load_wfrag(browg, srowg, k0, lane);
    v16h bu = load_wfrag(browu, srowu, k0, lane);
#pragma unroll
    for (int mt = 0; mt < 8; ++mt) {
      v16h a = load_afrag(&At[mt * 16 + ncol][khalf]);
      accg[mt] = __builtin_amdgcn_wmma_f32_16x16x32_f16(
          false, a, false, bg, (short)0, accg[mt], false, false);
      accu[mt] = __builtin_amdgcn_wmma_f32_16x16x32_f16(
          false, a, false, bu, (short)0, accu[mt], false, false);
    }
    wait_async0();     // next buffer's copies have landed (this wave)
    __syncthreads();   // ... and everyone finished reading the current buffer
  }

  const float bg_ = bias_gu[(size_t)e * 2 * FD + og];
  const float bu_ = bias_gu[(size_t)e * 2 * FD + ou];
  const int mbase = (lane >> 4) << 3;   // D: lanes 16-31 hold M = r + 8
#pragma unroll
  for (int mt = 0; mt < 8; ++mt) {
#pragma unroll
    for (int r = 0; r < 8; ++r) {
      float g = accg[mt][r] + bg_;
      float u = accu[mt][r] + bu_;
      float hv = (g / (1.0f + __expf(-g))) * u;   // silu(g) * u
      int t = mt * 16 + mbase + r;
      Hbuf[((size_t)e * NT + t) * FD + og] = (_Float16)hv;
    }
  }
}

// ---------------------------------------------------------------------------
// Kernel 4: y = h @ Wdown^T + bias; out += combine[t,e] * y  (atomic f32)
// grid (E, H/128); same double-buffered tiling as kernel 3.
// ---------------------------------------------------------------------------
__global__ __launch_bounds__(256) void moe_down_combine(
    const _Float16* __restrict__ Hbuf,
    const int* __restrict__ blocks_down,
    const int* __restrict__ scales_down,
    const float* __restrict__ bias_down,
    const float* __restrict__ combine,
    float* __restrict__ out) {
  __shared__ _Float16 Atile[2][NT][APAD];
  const int e     = blockIdx.x;
  const int htile = blockIdx.y * 128;
  const int wave  = threadIdx.x >> 5;
  const int lane  = threadIdx.x & 31;
  const int ncol  = lane & 15;
  const int khalf = (lane >> 4) << 3;
  const int oc = htile + wave * 16 + ncol;   // output hidden column
  const int* brow = blocks_down + ((size_t)e * HD + oc) * (FD / 2);
  const int* srow = scales_down + ((size_t)e * HD + oc) * (FD / 32);

  v8f acc[8] = {};
  const int srow_r = threadIdx.x >> 1;
  const int scol   = (threadIdx.x & 1) * 16;
  const uint32_t ldsoff0 = (uint32_t)(uintptr_t)&Atile[0][srow_r][scol];
  const uint32_t ldsoff1 = (uint32_t)(uintptr_t)&Atile[1][srow_r][scol];
  const _Float16* Asrc = Hbuf + (size_t)e * NT * FD + (size_t)srow_r * FD + scol;

  issue_async(ldsoff0, (const void*)Asrc);
  wait_async0();
  __syncthreads();

  const int NK = FD / 32;
  for (int i = 0; i < NK; ++i) {
    const int k0 = i * 32;
    if (i + 1 < NK)
      issue_async((i & 1) ? ldsoff0 : ldsoff1, (const void*)(Asrc + (i + 1) * 32));
    if (k0 + 32 < FD) __builtin_prefetch(brow + ((k0 + 32) >> 1), 0, 1);
    const _Float16 (*At)[APAD] = Atile[i & 1];
    v16h b = load_wfrag(brow, srow, k0, lane);
#pragma unroll
    for (int mt = 0; mt < 8; ++mt) {
      v16h a = load_afrag(&At[mt * 16 + ncol][khalf]);
      acc[mt] = __builtin_amdgcn_wmma_f32_16x16x32_f16(
          false, a, false, b, (short)0, acc[mt], false, false);
    }
    wait_async0();
    __syncthreads();
  }

  const float bd = bias_down[(size_t)e * HD + oc];
  const int mbase = (lane >> 4) << 3;
#pragma unroll
  for (int mt = 0; mt < 8; ++mt) {
#pragma unroll
    for (int r = 0; r < 8; ++r) {
      int t = mt * 16 + mbase + r;
      float w = combine[t * NE + e];          // 0 for non-top-2 experts
      float y = w * (acc[mt][r] + bd);
      atomicAdd(&out[(size_t)t * HD + oc], y);
    }
  }
}

// ---------------------------------------------------------------------------
extern "C" void kernel_launch(void* const* d_in, const int* in_sizes, int n_in,
                              void* d_out, int out_size, void* d_ws, size_t ws_size,
                              hipStream_t stream) {
  const float* hidden      = (const float*)d_in[0];  // [1,128,1024]
  const float* router_w    = (const float*)d_in[1];  // [16,1024]
  const float* bias_gu     = (const float*)d_in[2];  // [16,4096]
  const float* bias_down   = (const float*)d_in[3];  // [16,1024]
  const int*   blocks_gu   = (const int*)d_in[4];    // [16,4096,512]
  const int*   scales_gu   = (const int*)d_in[5];    // [16,4096,32]
  const int*   blocks_down = (const int*)d_in[6];    // [16,1024,1024]
  const int*   scales_down = (const int*)d_in[7];    // [16,1024,64]
  float* out = (float*)d_out;

  // workspace layout (all 16B aligned)
  char* ws = (char*)d_ws;
  float*    combine = (float*)ws;                          //   8 KB
  _Float16* Xh      = (_Float16*)(ws + 8192);              // 256 KB
  _Float16* Hbuf    = (_Float16*)(ws + 8192 + 262144);     // 8.4 MB (E*T*F f16)

  hipMemsetAsync(d_out, 0, (size_t)out_size * sizeof(float), stream);

  cvt_f32_f16<<<(NT * HD + 255) / 256, 256, 0, stream>>>(hidden, Xh, NT * HD);
  router_topk<<<NT, 256, 0, stream>>>(hidden, router_w, combine);
  moe_gu_swiglu<<<dim3(NE, FD / 128), 256, 0, stream>>>(
      Xh, blocks_gu, scales_gu, bias_gu, Hbuf);
  moe_down_combine<<<dim3(NE, HD / 128), 256, 0, stream>>>(
      Hbuf, blocks_down, scales_down, bias_down, combine, out);
}
